// Matching_layer_3917010174015
// MI455X (gfx1250) — matrix operand order repaired and checked by reading, not compiled, not run on hardware
//
#include <hip/hip_runtime.h>
#include <hip/hip_bf16.h>

typedef __attribute__((ext_vector_type(16))) _Float16 v16h;
typedef __attribute__((ext_vector_type(8)))  _Float16 v8h;
typedef __attribute__((ext_vector_type(8)))  float    v8f;

#define BB   4
#define CC   64
#define HH   64
#define WW   64
#define PP   9
#define PADH 72   // HH + 8
#define PADW 72   // WW + 8

// ---------------------------------------------------------------------------
// Zero-fill (for padded f2 staging buffer; covers the zero borders).
// ---------------------------------------------------------------------------
__global__ void zero_u32(uint32_t* __restrict__ p, int n) {
  int i = blockIdx.x * blockDim.x + threadIdx.x;
  if (i < n) p[i] = 0u;
}

// ---------------------------------------------------------------------------
// L2-normalize over channels and transpose NCHW fp32 -> [b][y][x][c] fp16.
// One block per (b, y). LDS tile [64 c][64 x] padded to 65 to avoid bank
// conflicts on the transposed read. Block-uniform bases come from blockIdx,
// so the compiler already keeps them scalar.
// ---------------------------------------------------------------------------
__global__ void norm_rows(const float* __restrict__ src, _Float16* __restrict__ dst,
                          int dH, int dW, int yoff, int xoff) {
  __shared__ float lds[64 * 65];
  __shared__ float rnorm[64];
  const int b = blockIdx.x >> 6;
  const int y = blockIdx.x & 63;
  const float* s = src + (size_t)b * CC * HH * WW + (size_t)y * WW; // [c][x] stride HH*WW
  for (int idx = threadIdx.x; idx < 64 * 64; idx += blockDim.x) {
    int c = idx >> 6, x = idx & 63;
    lds[c * 65 + x] = s[(size_t)c * HH * WW + x];   // coalesced along x
  }
  __syncthreads();
  if (threadIdx.x < 64) {
    int x = threadIdx.x;
    float acc = 0.f;
    #pragma unroll 8
    for (int c = 0; c < 64; ++c) { float v = lds[c * 65 + x]; acc += v * v; }
    rnorm[x] = 1.0f / sqrtf(acc + 1e-6f);
  }
  __syncthreads();
  _Float16* d = dst + (((size_t)b * dH + (y + yoff)) * dW + xoff) * CC;
  for (int idx = threadIdx.x; idx < 64 * 64; idx += blockDim.x) {
    int c = idx & 63, x = idx >> 6;                  // c fast -> coalesced f16 writes
    d[(size_t)x * CC + c] = (_Float16)(lds[c * 65 + x] * rnorm[x]);
  }
}

// ---------------------------------------------------------------------------
// Cost volume via banded GEMM on WMMA.
// One wave per (b, y, x0 in {0,16,32,48}, i=dy-index in [0,9)).
// A: 16 pixels x 64 ch of f1.  B: 32 shifted columns of the padded f2 row.
// Two 16x16 N-tiles, K=64 -> 2x v_wmma_f32_16x16x32_f16 each.
// Valid outputs (diagonals n_full = m + j) are staged through wave-private
// LDS so global stores are coalesced 64B rows per j.
// ---------------------------------------------------------------------------
#define DROW 27   // LDS row stride: conflict-free for both write and m*28+j read

__global__ void corr_wmma(const _Float16* __restrict__ f1h,
                          const _Float16* __restrict__ f2h,
                          float* __restrict__ out) {
  __shared__ float dtile[8][16 * DROW];
  const int wv   = threadIdx.x >> 5;
  const int lane = threadIdx.x & 31;
  const int wave = blockIdx.x * 8 + wv;

  int i = wave % 9;                 // dy = i - 4
  int t = wave / 9;
  const int x0 = (t & 3) << 4; t >>= 2;
  const int y  = t & 63;
  const int b  = t >> 6;

  const int lo16  = lane & 15;
  const int higrp = lane >> 4;      // 0 or 1
  const int mbase = higrp * 8;

  // Wave-uniform element offsets, forced into SGPRs so all global accesses
  // use the saddr + 32-bit voffset form.
  const int aoff = __builtin_amdgcn_readfirstlane(((b * HH + y) * WW + x0) * CC);
  const int boff = __builtin_amdgcn_readfirstlane(((b * PADH + (y + i)) * PADW + x0) * CC);
  const int ooff = __builtin_amdgcn_readfirstlane((b * 81 + i * 9) * (HH * WW) + y * WW + x0);
  const int ncap = __builtin_amdgcn_readfirstlane((PADW - 1) - x0);  // last valid column
  const _Float16* abase = f1h + aoff;
  const _Float16* bbase = f2h + boff;
  float*          obase = out + ooff;

  // ---- A: pixel x0+lo16; lanes 0-15: K={k..k+7, k+16..k+23}, lanes 16-31: +8
  const int alane = lo16 * CC + mbase;     // cbase == mbase (0 or 8)
  v8h t0 = *(const v8h*)(abase + alane);
  v8h t1 = *(const v8h*)(abase + alane + 16);
  v16h a0 = __builtin_shufflevector(t0, t1, 0,1,2,3,4,5,6,7,8,9,10,11,12,13,14,15);
  t0 = *(const v8h*)(abase + alane + 32);
  t1 = *(const v8h*)(abase + alane + 48);
  v16h a1 = __builtin_shufflevector(t0, t1, 0,1,2,3,4,5,6,7,8,9,10,11,12,13,14,15);

  float* dt = &dtile[wv][0];

  #pragma unroll
  for (int nb = 0; nb < 32; nb += 16) {
    // ---- B: column n = lo16 of this N-tile = padded-x (x0 + nb + n).
    // Columns past the padded row (n_full >= 24, never consumed by the
    // diagonal extraction) are branch-lessly clamped to the last valid
    // column, so every lane issues an unconditional saddr-form load.
    const int ncl = min(nb + lo16, ncap);
    const int bl  = ncl * CC + higrp * 8;   // lanes 0-15: K={0..7,16..23}, 16-31: +8
    v8h u0 = *(const v8h*)(bbase + bl);
    v8h u1 = *(const v8h*)(bbase + bl + 16);
    v16h b0 = __builtin_shufflevector(u0, u1, 0,1,2,3,4,5,6,7,8,9,10,11,12,13,14,15);
    u0 = *(const v8h*)(bbase + bl + 32);
    u1 = *(const v8h*)(bbase + bl + 48);
    v16h b1 = __builtin_shufflevector(u0, u1, 0,1,2,3,4,5,6,7,8,9,10,11,12,13,14,15);

    v8f acc = {};
    acc = __builtin_amdgcn_wmma_f32_16x16x32_f16(false, a0, false, b0,
                                                 (short)0, acc, false, false);
    acc = __builtin_amdgcn_wmma_f32_16x16x32_f16(false, a1, false, b1,
                                                 (short)0, acc, false, false);

    // ---- stage D tile rows in wave-private LDS: lane n = lo16, VGPR r -> M.
    const int nf = nb + lo16;
    #pragma unroll
    for (int r = 0; r < 8; ++r)
      dt[(mbase + r) * DROW + nf] = acc[r];
  }

  // Same-wave LDS ops are in-order: no barrier needed for wave-private data.
  // Diagonal read (addr = m*(DROW+1)+j is conflict-free), coalesced stores:
  // lanes 0-15 write 64B row for j, lanes 16-31 for j+1.
  #pragma unroll
  for (int round = 0; round < 5; ++round) {
    const int j = round * 2 + higrp;        // dx index
    const int m = lo16;
    if (j < PP) {
      const float v = dt[m * (DROW + 1) + j];
      obase[j * (HH * WW) + m] = fmaxf(v, 0.0f);
    }
  }
}

// ---------------------------------------------------------------------------
extern "C" void kernel_launch(void* const* d_in, const int* in_sizes, int n_in,
                              void* d_out, int out_size, void* d_ws, size_t ws_size,
                              hipStream_t stream) {
  (void)in_sizes; (void)n_in; (void)out_size; (void)ws_size;
  const float* f1 = (const float*)d_in[0];
  const float* f2 = (const float*)d_in[1];
  float* out = (float*)d_out;

  // Workspace layout: f1h [4][64][64][64] f16 (2 MiB), then f2h [4][72][72][64] f16.
  _Float16* f1h = (_Float16*)d_ws;
  _Float16* f2h = (_Float16*)((char*)d_ws + (size_t)BB * HH * WW * CC * sizeof(_Float16));

  // 1) zero padded f2 staging (borders stay zero => implicit OOB handling)
  const int nzero = (BB * PADH * PADW * CC) / 2;   // as dwords
  zero_u32<<<(nzero + 255) / 256, 256, 0, stream>>>((uint32_t*)f2h, nzero);

  // 2) normalize + transpose + f16 convert
  norm_rows<<<BB * HH, 256, 0, stream>>>(f1, f1h, HH, WW, 0, 0);
  norm_rows<<<BB * HH, 256, 0, stream>>>(f2, f2h, PADH, PADW, 4, 4);

  // 3) cost volume: 4*64*4*9 = 9216 waves, 8 waves (256 thr) per block
  corr_wmma<<<(BB * HH * (WW / 16) * PP) / 8, 256, 0, stream>>>(f1h, f2h, out);
}
